// BlockCirculantLinear_66194035966009
// MI455X (gfx1250) — matrix operand order repaired and checked by reading, not compiled, not run on hardware
//
#include <hip/hip_runtime.h>

// Block-circulant linear as an f16-WMMA GEMM:
//   Y(8192x4096) = X(8192x4096) * W(4096x4096),  W[k=i*256+a, m=j*256+b] = c[j,i,(b-a)&255]
//
// v4: - B-side synthesized once per block as 8 rotated copies of each
//       reversed+doubled circulant row in LDS; every WMMA B-fragment is two
//       ALIGNED ds_load_b128 (rho = b&7, p0 = (a0 - 8*(b>>3)) & 255).
//     - A-tile (fp32) staged by GLOBAL_LOAD_ASYNC_TO_LDS_B128 (ASYNCcnt +
//       s_wait_asynccnt); builtin signature (v4i as1*, v4i as3*, Ii, Ii)
//       confirmed by the round-3 diagnostic. fp32->f16 convert at frag load.
//     - One barrier per K-step (double-buffered A).
// HBM traffic = read x + write y (~268MB @ 23.3TB/s ~= 11.5us floor); x fits
// in the 192MB L2 so per-N-tile re-reads are L2 hits; c (256KB) is L2-hot.

typedef __attribute__((ext_vector_type(16))) _Float16 v16h;
typedef __attribute__((ext_vector_type(8)))  _Float16 v8h;
typedef __attribute__((ext_vector_type(8)))  float    v8f;
typedef __attribute__((ext_vector_type(4)))  int      v4i;

#define AS1 __attribute__((address_space(1)))
#define AS3 __attribute__((address_space(3)))

#if __has_builtin(__builtin_amdgcn_global_load_async_to_lds_b128)
#define USE_ASYNC_A 1
#else
#define USE_ASYNC_A 0
#endif

namespace {

constexpr int kK = 4096;             // in-features  (16 blocks * 256)
constexpr int kM = 4096;             // out-features (16 blocks * 256)
constexpr int kN = 8192;             // samples      (4 * 2048)

constexpr int BM = 128, BN = 128, BK = 32;
constexpr int KIT = kK / BK;         // 128 K-steps

constexpr int ASF      = BM * BK;                      // 4096 f32 per A buffer (16KB)
constexpr int R8_ELEMS = 16 * 8 * 512;                 // 65536 f16 (128KB)
constexpr unsigned A_BYTES = 2u * ASF * sizeof(float); // 32KB (double buffered)
constexpr unsigned DYN_LDS = A_BYTES + R8_ELEMS * (unsigned)sizeof(_Float16); // 160KB

__device__ inline v8h cvt8(const float4 a, const float4 b) {
  v8h r;
  r[0] = (_Float16)a.x; r[1] = (_Float16)a.y; r[2] = (_Float16)a.z; r[3] = (_Float16)a.w;
  r[4] = (_Float16)b.x; r[5] = (_Float16)b.y; r[6] = (_Float16)b.z; r[7] = (_Float16)b.w;
  return r;
}

} // namespace

__global__ __launch_bounds__(256)
void bcirc_wmma_f16(const float* __restrict__ x,
                    const float* __restrict__ c,
                    const float* __restrict__ bias,
                    float* __restrict__ y)
{
  extern __shared__ char smem[];
  float*    As = (float*)smem;                  // [2][BM*BK] fp32, 32KB
  _Float16* R8 = (_Float16*)(smem + A_BYTES);   // [16][8][512] f16, 128KB

  const int t  = threadIdx.x;
  const int m0 = blockIdx.y * BM;
  const int n0 = blockIdx.x * BN;   // 128-aligned => single output block j per tile
  const int jb = n0 >> 8;

  const int lane = t & 31;
  const int wave = t >> 5;
  const int wm   = wave & 3;        // 4 row-groups of 32
  const int wn   = wave >> 2;       // 2 col-groups of 64
  const int hh   = lane >> 4;
  const int lm   = lane & 15;

  // ---- one-time init (rolled loop): R8[i][rho][p] = c[j,i,(rho-p)&255] ----
  {
    const float* cj = c + (jb << 12);            // c[j] : 16 x 256 floats (L2-hot)
    #pragma unroll 1
    for (int s = 0; s < 32; ++s) {               // 8192 8-elem segments / 256 threads
      const int seg = (s << 8) | t;
      const int i   = seg >> 9;
      const int rho = (seg >> 6) & 7;
      const int p0  = (seg & 63) << 3;
      const float* crow = cj + (i << 8);
      v8h h;
      #pragma unroll
      for (int e = 0; e < 8; ++e)
        h[e] = (_Float16)crow[(rho - p0 - e) & 255];
      *(v8h*)&R8[(((i << 3) | rho) << 9) + p0] = h;
    }
  }

  // ---- per-lane B-fragment constants (fixed across K-steps) ----
  int rho_u[4], beta_u[4];
  #pragma unroll
  for (int u = 0; u < 4; ++u) {
    const int b = (n0 + wn * 64 + u * 16 + lm) & 255;
    rho_u[u]  = b & 7;
    beta_u[u] = (b >> 3) << 3;      // 8*(b>>3)
  }

  v8f acc[2][4];
  #pragma unroll
  for (int s = 0; s < 2; ++s)
    #pragma unroll
    for (int u = 0; u < 4; ++u)
      #pragma unroll
      for (int e = 0; e < 8; ++e) acc[s][u][e] = 0.0f;

  // ---- A-tile staging: 256 threads x 16 floats (64B) per buffer ----
  const int fr = t >> 1;
  const int fk = (t & 1) * 16;
  const float* xrow = x + (size_t)(m0 + fr) * kK + fk;

  auto fillA = [&](int buf, int step) {
    const float* xp = xrow + step * BK;
#if USE_ASYNC_A
    // DMA straight into LDS; as(3) pointer built from the 32-bit LDS offset.
    const unsigned lds0 =
        __builtin_amdgcn_groupstaticsize() + (unsigned)(buf * ASF + fr * BK + fk) * 4u;
    #pragma unroll
    for (int q = 0; q < 4; ++q)
      __builtin_amdgcn_global_load_async_to_lds_b128(
          (AS1 v4i*)(unsigned long long)(xp + 4 * q),
          (AS3 v4i*)(lds0 + 16u * q), 0, 0);
#else
    const float4 f0 = ((const float4*)xp)[0];
    const float4 f1 = ((const float4*)xp)[1];
    const float4 f2 = ((const float4*)xp)[2];
    const float4 f3 = ((const float4*)xp)[3];
    float* dst = As + buf * ASF + fr * BK + fk;
    ((float4*)dst)[0] = f0;
    ((float4*)dst)[1] = f1;
    ((float4*)dst)[2] = f2;
    ((float4*)dst)[3] = f3;
#endif
    __builtin_prefetch(xp + BK, 0, 1);           // -> global_prefetch_b8
  };

  auto waitAsync = [&]() {
#if USE_ASYNC_A
 #if __has_builtin(__builtin_amdgcn_s_wait_asynccnt)
    __builtin_amdgcn_s_wait_asynccnt(0);
 #else
    asm volatile("s_wait_asynccnt 0x0" ::: "memory");
 #endif
#endif
  };

  fillA(0, 0);

  for (int kt = 0; kt < KIT; ++kt) {
    waitAsync();                      // my async fills of A[kt&1] landed in LDS
    __syncthreads();                  // all fills + R8 init visible; prior reads done
    if (kt + 1 < KIT) fillA((kt + 1) & 1, kt + 1);

    const float* Ab = As + (kt & 1) * ASF;

    // A fragments (fp32 LDS -> f16): K = [16h,16h+16)
    v16h af[2];
    #pragma unroll
    for (int s = 0; s < 2; ++s) {
      const float4* p = (const float4*)(Ab + (wm * 32 + s * 16 + lm) * BK);
      const v8h lo = cvt8(p[4 * hh],     p[4 * hh + 1]);
      const v8h hi = cvt8(p[4 * hh + 2], p[4 * hh + 3]);
      af[s] = __builtin_shufflevector(lo, hi,
                0, 1, 2, 3, 4, 5, 6, 7, 8, 9, 10, 11, 12, 13, 14, 15);
    }

    // B fragments straight from rotated circulant rows: two aligned b128 each
    const int ib = kt >> 3;                          // input block i
    const int a0 = ((kt & 7) << 5) + (hh << 4);      // K-phase within the block
    v16h bfr[4];
    #pragma unroll
    for (int u = 0; u < 4; ++u) {
      const int p0 = (a0 - beta_u[u]) & 255;         // multiple of 8 -> 16B aligned
      const v8h* p = (const v8h*)&R8[(((ib << 3) | rho_u[u]) << 9) + p0];
      bfr[u] = __builtin_shufflevector(p[0], p[1],
                0, 1, 2, 3, 4, 5, 6, 7, 8, 9, 10, 11, 12, 13, 14, 15);
    }

    #pragma unroll
    for (int s = 0; s < 2; ++s)
      #pragma unroll
      for (int u = 0; u < 4; ++u)
        acc[s][u] = __builtin_amdgcn_wmma_f32_16x16x32_f16(
            false, af[s], false, bfr[u], (short)0, acc[s][u], false, false);
  }

  // ---- epilogue: C/D layout (VGPR r, half h) -> row = r + 8h, col = lm ----
  #pragma unroll
  for (int s = 0; s < 2; ++s) {
    #pragma unroll
    for (int u = 0; u < 4; ++u) {
      const int col = n0 + wn * 64 + u * 16 + lm;
      const float bv = bias[col];
      #pragma unroll
      for (int r = 0; r < 8; ++r) {
        const int row = m0 + wm * 32 + s * 16 + 8 * hh + r;
        y[(size_t)row * kM + col] = acc[s][u][r] + bv;
      }
    }
  }
}

extern "C" void kernel_launch(void* const* d_in, const int* in_sizes, int n_in,
                              void* d_out, int out_size, void* d_ws, size_t ws_size,
                              hipStream_t stream) {
  const float* x    = (const float*)d_in[0];   // (4, 2048, 4096) fp32
  const float* c    = (const float*)d_in[1];   // (16, 16, 256)  fp32
  const float* bias = (const float*)d_in[2];   // (4096,)        fp32
  float*       y    = (float*)d_out;           // (4, 2048, 4096) fp32
  (void)in_sizes; (void)n_in; (void)out_size; (void)d_ws; (void)ws_size;

  dim3 grid(kM / BN, kN / BM);                 // (32, 64) = 2048 workgroups
  bcirc_wmma_f16<<<grid, 256, DYN_LDS, stream>>>(x, c, bias, y);
}
